// HeteroCellBipartite_52544629899911
// MI455X (gfx1250) — matrix operand order repaired and checked by reading, not compiled, not run on hardware
//
#include <hip/hip_runtime.h>
#include <hip/hip_bf16.h>

#define HD   256
#define GNN  6607
#define RNN  4881
#define MNN  2534
#define EPN  200000
#define ERN  200000
#define EGN  40000
#define EMN  40000

typedef __attribute__((ext_vector_type(16))) __bf16 v16bf;
typedef __attribute__((ext_vector_type(8)))  float  v8f;

static __device__ __forceinline__ void atomic_max_f32(float* addr, float val) {
  if (val >= 0.0f) atomicMax((int*)addr, __float_as_int(val));
  else             atomicMin((unsigned int*)addr, (unsigned int)__float_as_int(val));
}

// ---------------------------------------------------------------------------
// Transpose-convert W[K=HD, N] f32 -> WT[N, K=HD] bf16 (one-shot per GEMM;
// weight reused by ~M/16 row tiles, so this makes B fragments contiguous).
// ---------------------------------------------------------------------------
__global__ __launch_bounds__(256)
void convert_wT_k(const float* __restrict__ W, __bf16* __restrict__ WT, int N) {
  int i = blockIdx.x * blockDim.x + threadIdx.x;          // i = k*N + n (coalesced read)
  if (i < N * HD) {
    const int k = i / N, n = i - k * N;
    WT[(size_t)n * HD + k] = (__bf16)W[i];
  }
}

// ---------------------------------------------------------------------------
// C[M,N] = A[M,HD] @ W[HD,N] + bias[N]  (optional fused ReLU)
// One wave computes a 16x64 tile: A fragment (f32->bf16 in-register, b64
// loads) reused across 4 WMMAs; B fragments are contiguous bf16 vector loads
// from the transposed weight copy. v_wmma_f32_16x16x32_bf16, f32 accum.
// Epilogue: wave-uniform fast path for full tiles (unguarded store clause).
// ---------------------------------------------------------------------------
__global__ __launch_bounds__(32)
void gemm_bias_wmma_k(const float* __restrict__ A, const __bf16* __restrict__ WT,
                      const float* __restrict__ bias, float* __restrict__ C,
                      int M, int N, int relu)
{
  const int lane = threadIdx.x & 31;
  const int half = lane >> 4;
  const int l16  = lane & 15;
  const int tn   = blockIdx.x * 64;          // 4 x 16 N-tiles per wave
  const int tm   = blockIdx.y << 4;
  int arow = tm + l16;
  if (arow >= M) arow = M - 1;               // clamp: unconditional loads, stores guarded
  const float* __restrict__ Arow = A + (size_t)arow * HD;

  v8f acc[4];
#pragma unroll
  for (int t = 0; t < 4; ++t) acc[t] = (v8f){};

  for (int kb = 0; kb < HD; kb += 32) {
    v16bf a;
#pragma unroll
    for (int v = 0; v < 8; ++v) {
      const int kk = 2 * v + 16 * half;
      const float2 p = *(const float2*)(Arow + kb + kk);   // global_load_b64
      a[2 * v]     = (__bf16)p.x;
      a[2 * v + 1] = (__bf16)p.y;
    }
#pragma unroll
    for (int t = 0; t < 4; ++t) {
      const int col = tn + t * 16 + l16;
      // B layout: lane col, K = kb + 16*half + 0..15 contiguous -> 32B vector load
      const v16bf b = *(const v16bf*)(WT + (size_t)col * HD + kb + 16 * half);
      acc[t] = __builtin_amdgcn_wmma_f32_16x16x32_bf16(false, a, false, b,
                                                       (short)0, acc[t], false, false);
    }
  }

  const int rbase = tm + 8 * half;           // rows rbase..rbase+7 per lane-half
  if (tm + 16 <= M) {
    // ---- fast path: full tile, no per-row guards ----
#pragma unroll
    for (int t = 0; t < 4; ++t) {
      const int col = tn + t * 16 + l16;
      const float bv = bias ? bias[col] : 0.0f;
      float* __restrict__ cp = C + (size_t)rbase * N + col;
#pragma unroll
      for (int v = 0; v < 8; ++v) {
        float x = acc[t][v] + bv;
        if (relu) x = fmaxf(x, 0.0f);
        cp[(size_t)v * N] = x;
      }
    }
  } else {
    // ---- tail tile: per-row guard ----
#pragma unroll
    for (int t = 0; t < 4; ++t) {
      const int col = tn + t * 16 + l16;
      const float bv = bias ? bias[col] : 0.0f;
#pragma unroll
      for (int v = 0; v < 8; ++v) {
        const int r = rbase + v;
        if (r < M) {
          float x = acc[t][v] + bv;
          if (relu) x = fmaxf(x, 0.0f);
          C[(size_t)r * N + col] = x;
        }
      }
    }
  }
}

// ---------------------------------------------------------------------------
__global__ __launch_bounds__(256)
void fill_k(float* __restrict__ p, float v, int n) {
  int i = blockIdx.x * blockDim.x + threadIdx.x;
  if (i < n) p[i] = v;
}

__global__ __launch_bounds__(256)
void row_bias_init_k(float* __restrict__ out, const float* __restrict__ bias, int total) {
  int i = blockIdx.x * blockDim.x + threadIdx.x;
  if (i < total) out[i] = bias[i & (HD - 1)];
}

__global__ __launch_bounds__(256)
void add_k(const float* __restrict__ a, const float* __restrict__ b,
           float* __restrict__ o, int n) {
  int i = blockIdx.x * blockDim.x + threadIdx.x;
  if (i < n) o[i] = a[i] + b[i];
}

// LayerNorm over H=256 + ReLU. One 256-thread block per row.
__global__ __launch_bounds__(HD)
void ln_relu_k(const float* __restrict__ in, const float* __restrict__ g,
               const float* __restrict__ b, float* __restrict__ out) {
  __shared__ float sm[HD];
  const int row = blockIdx.x;
  const int h = threadIdx.x;
  const float x = in[(size_t)row * HD + h];
  sm[h] = x; __syncthreads();
  for (int s = HD / 2; s > 0; s >>= 1) { if (h < s) sm[h] += sm[h + s]; __syncthreads(); }
  const float mean = sm[0] * (1.0f / HD);
  __syncthreads();
  const float d = x - mean;
  sm[h] = d * d; __syncthreads();
  for (int s = HD / 2; s > 0; s >>= 1) { if (h < s) sm[h] += sm[h + s]; __syncthreads(); }
  const float var = sm[0] * (1.0f / HD);
  const float y = d * rsqrtf(var + 1e-5f) * g[h] + b[h];
  out[(size_t)row * HD + h] = fmaxf(y, 0.0f);
}

// GATv2 edge logits + running segment max. One wave32 per edge, float4 loads.
__global__ __launch_bounds__(256)
void edge_logit_k(const float* __restrict__ xl, const float* __restrict__ xr,
                  const int* __restrict__ src, const int* __restrict__ dst,
                  const float* __restrict__ att, const float* __restrict__ ea,
                  const float* __restrict__ We,
                  float* __restrict__ logit, float* __restrict__ mbuf, int E)
{
  const int e = (blockIdx.x * blockDim.x + threadIdx.x) >> 5;
  const int lane = threadIdx.x & 31;
  if (e >= E) return;
  const int s = src[e], d = dst[e];
  const float eav = ea ? ea[e] : 0.0f;
  const float4* __restrict__ xls = (const float4*)(xl + (size_t)s * HD);
  const float4* __restrict__ xrs = (const float4*)(xr + (size_t)d * HD);
  const float4* __restrict__ at4 = (const float4*)att;
  const float4* __restrict__ we4 = (const float4*)We;
  float acc = 0.0f;
#pragma unroll
  for (int q = lane; q < HD / 4; q += 32) {               // 2 iterations
    const float4 va = xls[q];
    const float4 vb = xrs[q];
    const float4 aw = at4[q];
    float4 v = make_float4(va.x + vb.x, va.y + vb.y, va.z + vb.z, va.w + vb.w);
    if (We) {
      const float4 w = we4[q];
      v.x += eav * w.x; v.y += eav * w.y; v.z += eav * w.z; v.w += eav * w.w;
    }
    v.x = (v.x > 0.0f) ? v.x : 0.2f * v.x;
    v.y = (v.y > 0.0f) ? v.y : 0.2f * v.y;
    v.z = (v.z > 0.0f) ? v.z : 0.2f * v.z;
    v.w = (v.w > 0.0f) ? v.w : 0.2f * v.w;
    acc += v.x * aw.x + v.y * aw.y + v.z * aw.z + v.w * aw.w;
  }
  for (int off = 16; off > 0; off >>= 1) acc += __shfl_xor(acc, off);
  if (lane == 0) {
    logit[e] = acc;
    atomic_max_f32(&mbuf[d], acc);
  }
}

__global__ __launch_bounds__(256)
void edge_exp_k(const float* __restrict__ logit, const int* __restrict__ dst,
                const float* __restrict__ mbuf, float* __restrict__ ex,
                float* __restrict__ den, int E)
{
  int e = blockIdx.x * blockDim.x + threadIdx.x;
  if (e >= E) return;
  const int d = dst[e];
  const float v = expf(logit[e] - mbuf[d]);
  ex[e] = v;
  atomicAdd(&den[d], v);
}

// out[dst] += xl[src] * alpha ; one wave per edge, vector gather + f32 atomics.
__global__ __launch_bounds__(256)
void edge_scatter_k(const float* __restrict__ xl, const int* __restrict__ src,
                    const int* __restrict__ dst, const float* __restrict__ ex,
                    const float* __restrict__ den, float* __restrict__ acc, int E)
{
  const int e = (blockIdx.x * blockDim.x + threadIdx.x) >> 5;
  const int lane = threadIdx.x & 31;
  if (e >= E) return;
  const int s = src[e], d = dst[e];
  const float alpha = ex[e] / (den[d] + 1e-16f);
  const float4* __restrict__ xls = (const float4*)(xl + (size_t)s * HD);
  float* __restrict__ ad = acc + (size_t)d * HD;
#pragma unroll
  for (int q = lane; q < HD / 4; q += 32) {
    const float4 v = xls[q];
    atomicAdd(&ad[4 * q + 0], v.x * alpha);
    atomicAdd(&ad[4 * q + 1], v.y * alpha);
    atomicAdd(&ad[4 * q + 2], v.z * alpha);
    atomicAdd(&ad[4 * q + 3], v.w * alpha);
  }
}

// out[m] = A[m,:K] . w + bias[0]  (gate MLP second layer, K=128)
__global__ __launch_bounds__(256)
void rowdot_k(const float* __restrict__ A, const float* __restrict__ w,
              const float* __restrict__ bias, float* __restrict__ out, int M, int K)
{
  const int m = (blockIdx.x * blockDim.x + threadIdx.x) >> 5;
  const int lane = threadIdx.x & 31;
  if (m >= M) return;
  float acc = 0.0f;
  for (int k = lane; k < K; k += 32) acc += A[(size_t)m * K + k] * w[k];
  for (int off = 16; off > 0; off >>= 1) acc += __shfl_xor(acc, off);
  if (lane == 0) out[m] = acc + bias[0];
}

__global__ void init_scal_k(float* scal) { scal[0] = -1e30f; scal[1] = 0.0f; }

__global__ __launch_bounds__(256)
void reduce_max_k(const float* __restrict__ x, int n, float* __restrict__ scal) {
  float v = -1e30f;
  for (int i = blockIdx.x * blockDim.x + threadIdx.x; i < n; i += gridDim.x * blockDim.x)
    v = fmaxf(v, x[i]);
  for (int off = 16; off > 0; off >>= 1) v = fmaxf(v, __shfl_xor(v, off));
  if ((threadIdx.x & 31) == 0) atomic_max_f32(scal, v);
}

__global__ __launch_bounds__(256)
void exp_sum_k(const float* __restrict__ gate, float* __restrict__ exn,
               float* __restrict__ scal, int n) {
  const float m = scal[0];
  float part = 0.0f;
  for (int i = blockIdx.x * blockDim.x + threadIdx.x; i < n; i += gridDim.x * blockDim.x) {
    const float e = expf(gate[i] - m);
    exn[i] = e;
    part += e;
  }
  for (int off = 16; off > 0; off >>= 1) part += __shfl_xor(part, off);
  if ((threadIdx.x & 31) == 0) atomicAdd(&scal[1], part);
}

__global__ __launch_bounds__(HD)
void pool_k(const float* __restrict__ t, const float* __restrict__ exn,
            const float* __restrict__ scal, float* __restrict__ pooled, int n) {
  const int h = threadIdx.x;
  float acc = 0.0f;
  for (int i = 0; i < n; ++i) acc += exn[i] * t[(size_t)i * HD + h];
  pooled[h] = acc / scal[1];
}

__global__ __launch_bounds__(32)
void head_k(const float* __restrict__ pooled, const float* __restrict__ W,
            const float* __restrict__ b, float* __restrict__ out) {
  const int j = threadIdx.x;
  if (j < 2) {
    float acc = 0.0f;
    for (int h = 0; h < HD; ++h) acc += pooled[h] * W[h * 2 + j];
    out[j] = acc + b[j];
  }
}

// ---------------------------------------------------------------------------
static inline int cdiv(int a, int b) { return (a + b - 1) / b; }

struct Scratch { float *XL, *XRD, *ACC, *LOGIT, *EX, *MB, *DEN; __bf16* WT; };

static void gemm(const float* A, const float* W, const float* bias, float* C,
                 int M, int N, int relu, __bf16* WT, hipStream_t stream) {
  convert_wT_k<<<cdiv(N * HD, 256), 256, 0, stream>>>(W, WT, N);
  gemm_bias_wmma_k<<<dim3(N / 64, cdiv(M, 16)), dim3(32), 0, stream>>>(A, WT, bias, C, M, N, relu);
}

static void run_conv(const float* xs, const float* xd, int Ns, int Nd, int E,
                     const int* src, const int* dst,
                     const float* Wl, const float* bl, const float* Wr, const float* br,
                     const float* att, const float* bias,
                     const float* We, const float* ea,
                     const float* lng, const float* lnb,
                     float* out, const Scratch& S, hipStream_t stream)
{
  gemm(xs, Wl, bl, S.XL, Ns, HD, 0, S.WT, stream);
  gemm(xd, Wr, br, S.XRD, Nd, HD, 0, S.WT, stream);
  fill_k<<<cdiv(Nd, 256), 256, 0, stream>>>(S.MB, -1e30f, Nd);
  fill_k<<<cdiv(Nd, 256), 256, 0, stream>>>(S.DEN, 0.0f, Nd);
  row_bias_init_k<<<cdiv(Nd * HD, 256), 256, 0, stream>>>(S.ACC, bias, Nd * HD);
  edge_logit_k<<<cdiv(E, 8), 256, 0, stream>>>(S.XL, S.XRD, src, dst, att, ea, We, S.LOGIT, S.MB, E);
  edge_exp_k<<<cdiv(E, 256), 256, 0, stream>>>(S.LOGIT, dst, S.MB, S.EX, S.DEN, E);
  edge_scatter_k<<<cdiv(E, 8), 256, 0, stream>>>(S.XL, src, dst, S.EX, S.DEN, S.ACC, E);
  ln_relu_k<<<Nd, HD, 0, stream>>>(S.ACC, lng, lnb, out);
}

extern "C" void kernel_launch(void* const* d_in, const int* in_sizes, int n_in,
                              void* d_out, int out_size, void* d_ws, size_t ws_size,
                              hipStream_t stream) {
  (void)in_sizes; (void)n_in; (void)out_size; (void)ws_size;
  // ---- input map (JAX pytree flatten: sorted dict keys) ----
  const float* ea_rmr = (const float*)d_in[0];
  const float* aggWg1 = (const float*)d_in[1];   // [256,128]
  const float* aggWg2 = (const float*)d_in[2];   // [128,1]
  const float* aggWt  = (const float*)d_in[3];   // [256,256]
  const float* aggbg1 = (const float*)d_in[4];
  const float* aggbg2 = (const float*)d_in[5];
  const float* aggbt  = (const float*)d_in[6];
  const float* emb_g  = (const float*)d_in[7];
  const float* emb_m  = (const float*)d_in[8];
  const float* emb_r  = (const float*)d_in[9];
  const float* headW  = (const float*)d_in[10];
  const float* headb  = (const float*)d_in[11];
  // layers at 12 + L*33 ; pre at 78..83 ; edge indices 84..87
  const float* preW1 = (const float*)d_in[78];
  const float* preW2 = (const float*)d_in[79];
  const float* preB  = (const float*)d_in[80];
  const float* preb1 = (const float*)d_in[81];
  const float* preb2 = (const float*)d_in[82];
  const float* preG  = (const float*)d_in[83];
  const int* ei_phys = (const int*)d_in[84];
  const int* ei_reg  = (const int*)d_in[85];
  const int* ei_gpr  = (const int*)d_in[86];
  const int* ei_rmr  = (const int*)d_in[87];

  // ---- workspace layout (floats) ----
  float* ws = (float*)d_ws;
  size_t off = 0;
  auto take = [&](size_t n) { float* p = ws + off; off += n; return p; };
  const size_t GH = (size_t)GNN * HD, RH = (size_t)RNN * HD, MH = (size_t)MNN * HD;
  float* XG  = take(GH);
  float* XR  = take(RH);
  float* XM  = take(MH);
  float* O1  = take(GH);   // phys out / t
  float* O2  = take(GH);   // reg out
  float* ORr = take(RH);   // gpr out
  float* OMm = take(MH);   // rmr out
  Scratch S;
  S.XL    = take(GH);      // also gate1 [GN,128]
  S.XRD   = take(GH);
  S.ACC   = take(GH);
  S.WT    = (__bf16*)take((size_t)HD * HD / 2 + 64);  // bf16 transposed weight copy
  S.LOGIT = take(EPN);
  S.EX    = take(EPN);
  S.MB    = take(GNN);     // also gate
  S.DEN   = take(GNN);     // also exn
  float* SCAL   = take(8);
  float* POOLED = take(HD);

  // ---- PreProcessor: two (Linear -> LN -> ReLU), shared LN params ----
  gemm(emb_g, preW1, preb1, S.ACC, GNN, HD, 0, S.WT, stream);
  ln_relu_k<<<GNN, HD, 0, stream>>>(S.ACC, preG, preB, XG);
  gemm(XG, preW2, preb2, S.ACC, GNN, HD, 0, S.WT, stream);
  ln_relu_k<<<GNN, HD, 0, stream>>>(S.ACC, preG, preB, XG);

  const float* xg_p = XG;
  const float* xr_p = emb_r;
  const float* xm_p = emb_m;

  for (int L = 0; L < 2; ++L) {
    const int pb = 12 + L * 33;
    auto P = [&](int i) { return (const float*)d_in[pb + i]; };
    // offsets inside layer: gpr gat 0..5, ln 6..13 (gpr,phys,reg,rmr g/b),
    // phys 14..19, reg 20..25, rmr 26..32 (We first)
    run_conv(xg_p, xg_p, GNN, GNN, EPN, ei_phys, ei_phys + EPN,
             P(14), P(18), P(15), P(19), P(16), P(17), nullptr, nullptr,
             P(8), P(9), O1, S, stream);
    run_conv(xg_p, xg_p, GNN, GNN, ERN, ei_reg, ei_reg + ERN,
             P(20), P(24), P(21), P(25), P(22), P(23), nullptr, nullptr,
             P(10), P(11), O2, S, stream);
    run_conv(xg_p, xr_p, GNN, RNN, EGN, ei_gpr, ei_gpr + EGN,
             P(0), P(4), P(1), P(5), P(2), P(3), nullptr, nullptr,
             P(6), P(7), ORr, S, stream);
    run_conv(xr_p, xm_p, RNN, MNN, EMN, ei_rmr, ei_rmr + EMN,
             P(27), P(31), P(28), P(32), P(29), P(30), P(26), ea_rmr,
             P(12), P(13), OMm, S, stream);
    add_k<<<cdiv((int)GH, 256), 256, 0, stream>>>(O1, O2, XG, (int)GH);
    hipMemcpyAsync(XR, ORr, RH * sizeof(float), hipMemcpyDeviceToDevice, stream);
    hipMemcpyAsync(XM, OMm, MH * sizeof(float), hipMemcpyDeviceToDevice, stream);
    xg_p = XG; xr_p = XR; xm_p = XM;
  }

  // ---- AttentionalAggregation over genes ----
  gemm(XG, aggWg1, aggbg1, S.XL, GNN, 128, 1, S.WT, stream);
  rowdot_k<<<cdiv(GNN, 8), 256, 0, stream>>>(S.XL, aggWg2, aggbg2, S.MB, GNN, 128);
  gemm(XG, aggWt, aggbt, O1, GNN, HD, 1, S.WT, stream);
  init_scal_k<<<1, 1, 0, stream>>>(SCAL);
  reduce_max_k<<<cdiv(GNN, 256), 256, 0, stream>>>(S.MB, GNN, SCAL);
  exp_sum_k<<<cdiv(GNN, 256), 256, 0, stream>>>(S.MB, S.DEN, SCAL, GNN);
  pool_k<<<1, HD, 0, stream>>>(O1, S.DEN, SCAL, POOLED, GNN);
  head_k<<<1, 32, 0, stream>>>(POOLED, headW, headb, (float*)d_out);
}
